// Dual_GCN_63204738728397
// MI455X (gfx1250) — matrix operand order repaired and checked by reading, not compiled, not run on hardware
//
#include <hip/hip_runtime.h>
#include <hip/hip_bf16.h>
#include <cstdint>

// ---------------- problem constants ----------------
#define NN    50000        // nodes
#define NE    800000       // edges
#define NF    500          // input features
#define NH    256          // hidden
#define NC    10           // classes
#define KPAD  512          // NF padded to multiple of 64
#define NCOL  512          // combined hidden cols (both GCNs)
#define MT    3125         // NN / 16 (exact)

typedef unsigned short u16;
typedef __attribute__((ext_vector_type(8)))  u16    u16x8;
typedef __attribute__((ext_vector_type(16))) u16    u16x16;
typedef __attribute__((ext_vector_type(16))) __bf16 v16bf;
typedef __attribute__((ext_vector_type(8)))  float  v8f;

union FragU { u16x16 u; v16bf b; };

// f32 -> bf16, round to nearest even
__device__ __forceinline__ u16 f2bf(float f) {
    unsigned u = __float_as_uint(f);
    unsigned r = u + 0x7FFFu + ((u >> 16) & 1u);
    return (u16)(r >> 16);
}
__device__ __forceinline__ unsigned pack2bf(float a, float b) {
    return (unsigned)f2bf(a) | ((unsigned)f2bf(b) << 16);
}
__device__ __forceinline__ float bf2f(unsigned v) {
    return __uint_as_float(v << 16);
}

// ---------------- zero fill ----------------
__global__ void zero4_kernel(float4* __restrict__ p, int n4) {
    int i = blockIdx.x * blockDim.x + threadIdx.x;
    if (i < n4) p[i] = make_float4(0.f, 0.f, 0.f, 0.f);
}

// ---------------- pack W1_1 | W1_2 into WMMA B-fragment layout (bf16) --------
// Bp1[((kt*32 + nt)*32 + lane)*16 + j] :
//   column n = nt*16 + (lane&15), K = kt*32 + (lane>=16 ? 8 : 0) + j + (j>=8 ? 8 : 0)
__global__ void packB1_kernel(const float* __restrict__ W11,
                              const float* __restrict__ W12,
                              u16* __restrict__ Bp1) {
    int tid = blockIdx.x * blockDim.x + threadIdx.x;     // 16*32*32 = 16384
    if (tid >= 16 * 32 * 32) return;
    int lane = tid & 31;
    int nt   = (tid >> 5) & 31;
    int kt   = tid >> 10;
    int n    = nt * 16 + (lane & 15);
    int kb   = kt * 32 + ((lane >> 4) << 3);
#pragma unroll
    for (int j = 0; j < 16; ++j) {
        int k = kb + j + (j >= 8 ? 8 : 0);
        float v = 0.f;
        if (k < NF) v = (n < NH) ? W11[(size_t)k * NH + n]
                                 : W12[(size_t)k * NH + (n - NH)];
        Bp1[(size_t)tid * 16 + j] = f2bf(v);
    }
}

// ---------------- pack W2_1 / W2_2 (K=256, N padded 10->16) -----------------
__global__ void packB2_kernel(const float* __restrict__ W21,
                              const float* __restrict__ W22,
                              u16* __restrict__ Bp2) {
    int tid = blockIdx.x * blockDim.x + threadIdx.x;     // 2*8*32 = 512
    if (tid >= 2 * 8 * 32) return;
    int lane = tid & 31;
    int kt   = (tid >> 5) & 7;
    int g    = tid >> 8;
    const float* W = g ? W22 : W21;
    int n  = lane & 15;
    int kb = kt * 32 + ((lane >> 4) << 3);
#pragma unroll
    for (int j = 0; j < 16; ++j) {
        int k = kb + j + (j >= 8 ? 8 : 0);
        float v = (n < NC) ? W[(size_t)k * NC + n] : 0.f;
        Bp2[(size_t)tid * 16 + j] = f2bf(v);
    }
}

// ---------------- fused dual linear1: H = bf16(x @ [W1_1|W1_2] + b1) --------
// grid (3125), block 256 (8 waves). Each block covers all 512 output cols:
// wave w owns N-tiles w*4 .. w*4+3 (4 accumulators, A-fragment reused 4x).
// x is streamed exactly once; K staged in double-buffered 16x64 bf16 LDS tiles.
__global__ void gemm1_kernel(const float* __restrict__ x,
                             const float* __restrict__ b11,
                             const float* __restrict__ b12,
                             const u16*   __restrict__ Bp1,
                             u16*         __restrict__ H) {
    __shared__ u16 Abuf[2][16 * 64];               // double-buffered A tiles
    const int t    = threadIdx.x;
    const int wv   = t >> 5;
    const int lane = t & 31;
    const int m0   = blockIdx.x * 16;

    const int rowl = lane & 15;
    const int kb   = (lane >> 4) << 3;             // 0 or 8

    // staging: thread t loads one float4 of x -> 4 bf16 (8B) into LDS
    const int srr = t >> 4;                        // row 0..15
    const int scc = (t & 15) << 2;                 // col-in-chunk 0,4,..,60
    const float* xrow = x + (size_t)(m0 + srr) * NF;

    v8f acc[4];
    const v8f vzero = {0.f, 0.f, 0.f, 0.f, 0.f, 0.f, 0.f, 0.f};
#pragma unroll
    for (int q = 0; q < 4; ++q) acc[q] = vzero;

    // ---- prologue: stage K-chunk 0 ----
    {
        int col = scc;                             // kc = 0
        float4 v = make_float4(0.f, 0.f, 0.f, 0.f);
        if (col < NF) v = *(const float4*)(xrow + col);
        uint2 pk;
        pk.x = pack2bf(v.x, v.y);
        pk.y = pack2bf(v.z, v.w);
        *(uint2*)&Abuf[0][srr * 64 + scc] = pk;
    }

    for (int kc = 0; kc < KPAD / 64; ++kc) {
        // ---- prefetch next K-chunk of x into registers ----
        uint2 nxt;
        nxt.x = 0u; nxt.y = 0u;
        if (kc + 1 < KPAD / 64) {
            int col = (kc + 1) * 64 + scc;
            float4 v = make_float4(0.f, 0.f, 0.f, 0.f);
            if (col < NF) v = *(const float4*)(xrow + col);
            nxt.x = pack2bf(v.x, v.y);
            nxt.y = pack2bf(v.z, v.w);
        }
        __syncthreads();                           // Abuf[kc&1] ready

        const u16* Acur = Abuf[kc & 1];
#pragma unroll
        for (int s = 0; s < 2; ++s) {              // two K=32 sub-steps
            u16x8 lo = *(const u16x8*)&Acur[rowl * 64 + s * 32 + kb];
            u16x8 hi = *(const u16x8*)&Acur[rowl * 64 + s * 32 + kb + 16];
            FragU a;
            a.u = __builtin_shufflevector(lo, hi, 0, 1, 2, 3, 4, 5, 6, 7,
                                          8, 9, 10, 11, 12, 13, 14, 15);
            const int ktb = kc * 2 + s;            // B k-tile index 0..15
#pragma unroll
            for (int q = 0; q < 4; ++q) {
                const int ntile = wv * 4 + q;
                FragU bf;
                bf.u = *(const u16x16*)(Bp1 +
                         ((size_t)(ktb * 32 + ntile) * 32 + lane) * 16);
                acc[q] = __builtin_amdgcn_wmma_f32_16x16x32_bf16(
                    false, a.b, false, bf.b, (short)0, acc[q], false, false);
            }
        }
        // ---- write prefetched chunk into the other buffer ----
        if (kc + 1 < KPAD / 64)
            *(uint2*)&Abuf[(kc + 1) & 1][srr * 64 + scc] = nxt;
    }

    // C layout: VGPR r -> M = r (lanes 0-15) / 8+r (lanes 16-31), N = lane&15
    const int nl  = lane & 15;
    const int mr0 = m0 + ((lane >> 4) << 3);
#pragma unroll
    for (int q = 0; q < 4; ++q) {
        const int ncol = (wv * 4 + q) * 16 + nl;
        const float bias = (ncol < NH) ? b11[ncol] : b12[ncol - NH];
#pragma unroll
        for (int r = 0; r < 8; ++r)
            H[(size_t)(mr0 + r) * NCOL + ncol] = f2bf(acc[q][r] + bias);
    }
}

// ---------------- spmm1: S[row] += w * H[col]  (both GCNs, 512 cols) --------
// thread = (edge, 4-col group); gather bf16, scatter f32 atomics
__global__ void spmm1_kernel(const int*   __restrict__ ei,
                             const float* __restrict__ ew,
                             const u16*   __restrict__ H,
                             float*       __restrict__ S) {
    int idx = blockIdx.x * blockDim.x + threadIdx.x;   // NE*128 threads
    int e = idx >> 7;
    int q = (idx & 127) << 2;                          // col group *4
    if (e >= NE) return;
    int   row = ei[e];
    int   col = ei[NE + e];
    float w   = ew[e];

    uint2 pk = *(const uint2*)(H + (size_t)col * NCOL + q);
    float f0 = bf2f(pk.x & 0xFFFFu);
    float f1 = bf2f(pk.x >> 16);
    float f2 = bf2f(pk.y & 0xFFFFu);
    float f3 = bf2f(pk.y >> 16);

    float* sp = S + (size_t)row * NCOL + q;
    atomicAdd(sp + 0, w * f0);
    atomicAdd(sp + 1, w * f1);
    atomicAdd(sp + 2, w * f2);
    atomicAdd(sp + 3, w * f3);
}

// ---------------- gemm2: G[g] = relu(S[:, g*256:+256]) @ W2_g + b2_g --------
// grid (ceil(3125/8), 2), block 256. relu + bf16 convert fused into A-frag.
__global__ void gemm2_kernel(const float* __restrict__ S,
                             const float* __restrict__ b21,
                             const float* __restrict__ b22,
                             const u16*   __restrict__ Bp2,
                             float*       __restrict__ G) {
    const int wv    = threadIdx.x >> 5;
    const int lane  = threadIdx.x & 31;
    const int mtile = blockIdx.x * 8 + wv;
    const int g     = blockIdx.y;
    if (mtile >= MT) return;
    const int m0 = mtile * 16;
    const float* b2 = g ? b22 : b21;

    const int rowl = lane & 15;
    const int kb   = (lane >> 4) << 3;
    const float* Srow = S + (size_t)(m0 + rowl) * NCOL + g * NH;

    v8f acc = {0.f, 0.f, 0.f, 0.f, 0.f, 0.f, 0.f, 0.f};
#pragma unroll
    for (int kt = 0; kt < NH / 32; ++kt) {
        int base = kt * 32 + kb;
        float4 f0 = *(const float4*)(Srow + base);
        float4 f1 = *(const float4*)(Srow + base + 4);
        float4 f2 = *(const float4*)(Srow + base + 16);
        float4 f3 = *(const float4*)(Srow + base + 20);
        FragU a;
        a.u[0]  = f2bf(fmaxf(f0.x, 0.f)); a.u[1]  = f2bf(fmaxf(f0.y, 0.f));
        a.u[2]  = f2bf(fmaxf(f0.z, 0.f)); a.u[3]  = f2bf(fmaxf(f0.w, 0.f));
        a.u[4]  = f2bf(fmaxf(f1.x, 0.f)); a.u[5]  = f2bf(fmaxf(f1.y, 0.f));
        a.u[6]  = f2bf(fmaxf(f1.z, 0.f)); a.u[7]  = f2bf(fmaxf(f1.w, 0.f));
        a.u[8]  = f2bf(fmaxf(f2.x, 0.f)); a.u[9]  = f2bf(fmaxf(f2.y, 0.f));
        a.u[10] = f2bf(fmaxf(f2.z, 0.f)); a.u[11] = f2bf(fmaxf(f2.w, 0.f));
        a.u[12] = f2bf(fmaxf(f3.x, 0.f)); a.u[13] = f2bf(fmaxf(f3.y, 0.f));
        a.u[14] = f2bf(fmaxf(f3.z, 0.f)); a.u[15] = f2bf(fmaxf(f3.w, 0.f));

        FragU bf;
        bf.u = *(const u16x16*)(Bp2 + ((size_t)(g * 8 + kt) * 32 + lane) * 16);

        acc = __builtin_amdgcn_wmma_f32_16x16x32_bf16(
            false, a.b, false, bf.b, (short)0, acc, false, false);
    }

    const int nl = lane & 15;
    const float bias = (nl < NC) ? b2[nl] : 0.f;
    const int mr0 = m0 + ((lane >> 4) << 3);
#pragma unroll
    for (int r = 0; r < 8; ++r)
        G[((size_t)g * NN + (mr0 + r)) * 16 + nl] = acc[r] + bias;
}

// ---------------- spmm2: out[g][row] += w * G[g][col][0:10] -----------------
__global__ void spmm2_kernel(const int*   __restrict__ ei,
                             const float* __restrict__ ew,
                             const float* __restrict__ G,
                             float*       __restrict__ out) {
    int e = blockIdx.x * blockDim.x + threadIdx.x;
    if (e >= NE) return;
    int   row = ei[e];
    int   col = ei[NE + e];
    float w   = ew[e];
#pragma unroll
    for (int g = 0; g < 2; ++g) {
        const float* gp = G + ((size_t)g * NN + col) * 16;
        float*       op = out + (size_t)g * (NN * NC) + (size_t)row * NC;
#pragma unroll
        for (int n = 0; n < NC; ++n)
            atomicAdd(op + n, w * gp[n]);
    }
}

// ---------------- launcher ----------------
extern "C" void kernel_launch(void* const* d_in, const int* in_sizes, int n_in,
                              void* d_out, int out_size, void* d_ws, size_t ws_size,
                              hipStream_t stream) {
    const float* x   = (const float*)d_in[0];
    const int*   ei  = (const int*)  d_in[1];   // edge_index [2, NE]
    const float* ew  = (const float*)d_in[2];
    const float* W11 = (const float*)d_in[3];
    const float* b11 = (const float*)d_in[4];
    const float* W21 = (const float*)d_in[5];
    const float* b21 = (const float*)d_in[6];
    const float* W12 = (const float*)d_in[7];
    const float* b12 = (const float*)d_in[8];
    const float* W22 = (const float*)d_in[9];
    const float* b22 = (const float*)d_in[10];
    float* out = (float*)d_out;

    // workspace layout (bytes, all 32B aligned)
    char* ws = (char*)d_ws;
    u16*   H   = (u16*)  (ws);                          //  51,200,000  bf16 hidden
    float* S   = (float*)(ws + 51200000);               // 102,400,000  spmm1 accum
    float* G   = (float*)(ws + 153600000);              //   6,400,000  linear2 out
    u16*   Bp1 = (u16*)  (ws + 160000000);              //     524,288  packed W1
    u16*   Bp2 = (u16*)  (ws + 160524288);              //      16,384  packed W2

    // 1) zero the atomic accumulators (S) and the output
    {
        int n4 = NN * NCOL / 4;                          // 6,400,000
        zero4_kernel<<<(n4 + 255) / 256, 256, 0, stream>>>((float4*)S, n4);
        int o4 = (2 * NN * NC) / 4;                      // 250,000
        zero4_kernel<<<(o4 + 255) / 256, 256, 0, stream>>>((float4*)out, o4);
    }
    // 2) pack weights into WMMA B-fragment bf16 layout
    packB1_kernel<<<(16 * 32 * 32 + 255) / 256, 256, 0, stream>>>(W11, W12, Bp1);
    packB2_kernel<<<2, 256, 0, stream>>>(W21, W22, Bp2);

    // 3) fused dual linear1 (WMMA bf16, f32 accum), x streamed once
    gemm1_kernel<<<MT, 256, 0, stream>>>(x, b11, b12, Bp1, H);

    // 4) spmm layer 1 (gather bf16 H, scatter f32 atomics)
    {
        long long work = (long long)NE * 128;
        spmm1_kernel<<<(unsigned)(work / 256), 256, 0, stream>>>(ei, ew, H, S);
    }

    // 5) relu + linear2 (WMMA, relu fused into A-fragment build)
    gemm2_kernel<<<dim3((MT + 7) / 8, 2), 256, 0, stream>>>(S, b21, b22, Bp2, G);

    // 6) spmm layer 2 -> d_out
    spmm2_kernel<<<(NE + 255) / 256, 256, 0, stream>>>(ei, ew, G, out);
}
